// MultiModalFusion_57140244906004
// MI455X (gfx1250) — matrix operand order repaired and checked by reading, not compiled, not run on hardware
//
#include <hip/hip_runtime.h>

typedef __attribute__((ext_vector_type(16))) __bf16 v16bf;
typedef __attribute__((ext_vector_type(2)))  __bf16 v2bf;
typedef __attribute__((ext_vector_type(2)))  float  v2f;
typedef __attribute__((ext_vector_type(8)))  float  v8f;
typedef __attribute__((ext_vector_type(8)))  unsigned short us8;
typedef __attribute__((ext_vector_type(4)))  unsigned u32x4;
typedef __attribute__((ext_vector_type(8)))  int i32x8;
typedef __attribute__((ext_vector_type(4)))  int i32x4;

union Frag16 {
  v16bf v;
  unsigned short s[16];
  unsigned w[8];
  us8 h[2];
};

union US8 { us8 v; unsigned short e[8]; };

__device__ __forceinline__ unsigned pk2bf(float lo, float hi) {
#if __has_builtin(__builtin_amdgcn_cvt_pk_bf16_f32)
  v2bf r = __builtin_amdgcn_cvt_pk_bf16_f32(lo, hi);
  return __builtin_bit_cast(unsigned, r);
#else
  v2f ff = {lo, hi};
  v2bf r = __builtin_convertvector(ff, v2bf);
  return __builtin_bit_cast(unsigned, r);
#endif
}

__device__ __forceinline__ unsigned short s2bf(float f) {
  return __builtin_bit_cast(unsigned short, (__bf16)f);
}

__device__ __forceinline__ float fast_tanh(float x) {
#if __has_builtin(__builtin_amdgcn_tanhf)
  return __builtin_amdgcn_tanhf(x);
#elif __has_builtin(__builtin_amdgcn_tanh_f32)
  return __builtin_amdgcn_tanh_f32(x);
#else
  return tanhf(x);
#endif
}

#define SBT 136   // LDS row stride (halfwords) for 128x128 B^T tile (272B rows)
#define SGT 40    // LDS row stride (halfwords) for 128x32 transposed tiles (80B rows)

// ---------------------------------------------------------------------------
// Stage a row-major 128x128 bf16 matrix into LDS with SBT-padded rows.
// Preferred path: single Tensor Data Mover DMA (tensor_load_to_lds) issued by
// wave 0, using the D# pad feature: 4 DWORDs of pad per 64-DWORD (256B) row
// -> 272B LDS row stride == SBT. Builtin arity differs per toolchain:
// clang>=23 takes 6 args, clang-22 (ROCm 7.2) takes 5.
// Fallback: cooperative b128 copy. Block must have 128 threads.
// ---------------------------------------------------------------------------
#if __has_builtin(__builtin_amdgcn_tensor_load_to_lds)
#define USE_TDM_STAGE 1
#endif

__device__ __forceinline__ void stage_bt(unsigned short* BT,
                                         const unsigned short* __restrict__ src,
                                         int tid) {
#if defined(USE_TDM_STAGE)
  if (tid < 32) {  // wave 0 issues one TDM DMA for the whole tile
    const unsigned long long ga = (unsigned long long)(size_t)src;
    const unsigned lds = (unsigned)(unsigned long long)
        ((__attribute__((address_space(3))) unsigned short*)BT);
    u32x4 g0;
    g0.x = 1u;                                                   // count=1 (valid)
    g0.y = lds;                                                  // lds_addr
    g0.z = (unsigned)ga;                                         // global_addr[31:0]
    g0.w = (unsigned)((ga >> 32) & 0x1FFFFFFull) | (2u << 30);   // addr[56:32] | type=2
    i32x8 g1;
    g1[0] = (1 << 16)      // data_size = 2 bytes
          | (1 << 20)      // pad_enable
          | (5 << 22)      // pad_interval: 64 DWORDs (one 256B row)
          | (3 << 25);     // pad_amount: 4 DWORDs (16B) -> 272B row stride
    g1[1] = 0x00800000;    // tensor_dim0 = 128  (bits 63:48)
    g1[2] = 0x00800000;    // tensor_dim1 = 128  (bits 95:80)
    g1[3] = 0x00800000;    // tile_dim0  = 128   (bits 127:112)
    g1[4] = 0x00000080;    // tile_dim1  = 128   (bits 143:128)
    g1[5] = 128;           // tensor_dim0_stride = 128 (bits 191:160)
    g1[6] = 0;
    g1[7] = 0;
    i32x4 z4 = {0, 0, 0, 0};
#if defined(__clang_major__) && (__clang_major__ >= 23)
    i32x8 z8 = {0, 0, 0, 0, 0, 0, 0, 0};
    __builtin_amdgcn_tensor_load_to_lds(g0, g1, z4, z4, z8, 0);
#else
    __builtin_amdgcn_tensor_load_to_lds(g0, g1, z4, z4, 0);
#endif
#if __has_builtin(__builtin_amdgcn_s_wait_tensorcnt)
    __builtin_amdgcn_s_wait_tensorcnt(0);
#endif
  }
#else
  for (int idx = tid; idx < 2048; idx += 128) {
    const int n = idx >> 4, k8 = (idx & 15) << 3;
    *(us8*)&BT[n * SBT + k8] = *(const us8*)(src + n * 128 + k8);
  }
#endif
  __syncthreads();
}

// A fragment (16x32 bf16) from row-major bf16 global row p (+kk applied by caller)
__device__ __forceinline__ void load_afrag_bf(Frag16& f, const unsigned short* __restrict__ p,
                                              int half) {
  f.h[0] = *(const us8*)(p + 16 * half);
  f.h[1] = *(const us8*)(p + 16 * half + 8);
}

// B fragment (32x16 bf16) from LDS row [n][k..]
__device__ __forceinline__ void load_bfrag(Frag16& f, const unsigned short* bp) {
  f.h[0] = *(const us8*)bp;
  f.h[1] = *(const us8*)(bp + 8);
}

__device__ __forceinline__ v8f wmma_bf16(const Frag16& a, const Frag16& b, v8f c) {
  return __builtin_amdgcn_wmma_f32_16x16x32_bf16(false, a.v, false, b.v, (short)0, c,
                                                 false, false);
}

// ---------------------------------------------------------------------------
__global__ __launch_bounds__(256)
void k_cvt_bf16(const float* __restrict__ src, unsigned* __restrict__ dst) {
  const size_t i = (size_t)blockIdx.x * 256 + threadIdx.x;
  dst[i] = pk2bf(src[2 * i], src[2 * i + 1]);
}

__global__ __launch_bounds__(256)
void k_wt_bf16(const float* __restrict__ W, unsigned* __restrict__ WT) {
  const int mat = blockIdx.x;
  const float* Wm = W + (size_t)mat * 16384;
  unsigned* WTm = WT + (size_t)mat * 8192;
  for (int j = threadIdx.x; j < 8192; j += 256) {
    const int n = j >> 6, k = (j & 63) << 1;
    WTm[n * 64 + (j & 63)] = pk2bf(Wm[(size_t)k * 128 + n], Wm[(size_t)(k + 1) * 128 + n]);
  }
}

// ---------------------------------------------------------------------------
// C[M,128] = act(A_bf[M,128] @ W + bias); W pre-transposed bf16 WT[n][k].
// 128 threads = 4 waves, two 16-row bands per wave. grid.x = M/128.
// ---------------------------------------------------------------------------
__global__ __launch_bounds__(128)
void k_gemm_proj(const unsigned short* __restrict__ A, const unsigned short* __restrict__ WT,
                 const float* __restrict__ bias, float* __restrict__ outF,
                 unsigned short* __restrict__ outBF, int do_tanh) {
  __shared__ unsigned short BT[128 * SBT];
  const int tid = threadIdx.x;
  const int lane = tid & 31, wave = tid >> 5;
  const int half = lane >> 4, l15 = lane & 15;

  stage_bt(BT, WT, tid);

  const int row0 = blockIdx.x * 128 + wave * 16;
  const unsigned short* arow0 = A + (size_t)(row0 + l15) * 128;
  const unsigned short* arow1 = arow0 + 64 * 128;

  v8f acc0[8] = {}, acc1[8] = {};
#pragma unroll
  for (int kk = 0; kk < 128; kk += 32) {
    Frag16 af0, af1;
    load_afrag_bf(af0, arow0 + kk, half);
    load_afrag_bf(af1, arow1 + kk, half);
    Frag16 bfr[2];
    load_bfrag(bfr[0], &BT[l15 * SBT + kk + 16 * half]);
#pragma unroll
    for (int t = 0; t < 8; ++t) {
      if (t < 7) load_bfrag(bfr[(t + 1) & 1], &BT[((t + 1) * 16 + l15) * SBT + kk + 16 * half]);
      acc0[t] = wmma_bf16(af0, bfr[t & 1], acc0[t]);
      acc1[t] = wmma_bf16(af1, bfr[t & 1], acc1[t]);
    }
  }
#pragma unroll
  for (int t = 0; t < 8; ++t) {
    const int col = t * 16 + l15;
    const float bv = bias ? bias[col] : 0.f;
#pragma unroll
    for (int v = 0; v < 8; ++v) {
      const int m = row0 + half * 8 + v;
      float v0 = acc0[t][v] + bv;
      float v1 = acc1[t][v] + bv;
      if (do_tanh) { v0 = fast_tanh(v0); v1 = fast_tanh(v1); }
      if (outF) {
        outF[(size_t)m * 128 + col] = v0;
        outF[(size_t)(m + 64) * 128 + col] = v1;
      }
      if (outBF) {
        outBF[(size_t)m * 128 + col] = s2bf(v0);
        outBF[(size_t)(m + 64) * 128 + col] = s2bf(v1);
      }
    }
  }
}

// ---------------------------------------------------------------------------
// Partial Gram: Gpart[b,chunk] = X_bf[kchunk]^T @ diag(mask) @ Y_bf[kchunk].
// ---------------------------------------------------------------------------
__global__ __launch_bounds__(256)
void k_gemm_gram(const unsigned short* __restrict__ X, const unsigned short* __restrict__ Y,
                 const int* __restrict__ wmask, float* __restrict__ Gpart,
                 int L, int chunkLen) {
  __shared__ unsigned short XT[128 * SGT];
  __shared__ unsigned short YT[128 * SGT];
  const int b = blockIdx.x, chunk = blockIdx.y;
  const int tid = threadIdx.x, lane = tid & 31, wave = tid >> 5;
  const int half = lane >> 4, l15 = lane & 15;
  const size_t base = (size_t)b * L * 128;
  const int k0beg = chunk * chunkLen, k0end = k0beg + chunkLen;
  const int q = (tid >> 4) << 1;
  const int c8 = (tid & 15) << 3;

  v8f acc[8] = {};
  for (int k0 = k0beg; k0 < k0end; k0 += 32) {
    __syncthreads();
    {
      const int m0 = wmask[b * L + k0 + q];
      const int m1 = wmask[b * L + k0 + q + 1];
      US8 xa, xb, ya, yb;
      xa.v = *(const us8*)(X + base + (size_t)(k0 + q) * 128 + c8);
      xb.v = *(const us8*)(X + base + (size_t)(k0 + q + 1) * 128 + c8);
      ya.v = *(const us8*)(Y + base + (size_t)(k0 + q) * 128 + c8);
      yb.v = *(const us8*)(Y + base + (size_t)(k0 + q + 1) * 128 + c8);
#pragma unroll
      for (int j = 0; j < 8; ++j) {
        *(unsigned*)&XT[(c8 + j) * SGT + q] = (unsigned)xa.e[j] | ((unsigned)xb.e[j] << 16);
        const unsigned ylo = m0 ? (unsigned)ya.e[j] : 0u;
        const unsigned yhi = m1 ? (unsigned)yb.e[j] : 0u;
        *(unsigned*)&YT[(c8 + j) * SGT + q] = ylo | (yhi << 16);
      }
    }
    if (k0 + 32 < k0end) {
      __builtin_prefetch(X + base + (size_t)(k0 + 32 + q) * 128 + c8, 0, 1);
      __builtin_prefetch(Y + base + (size_t)(k0 + 32 + q) * 128 + c8, 0, 1);
    }
    __syncthreads();
    Frag16 af;
    const unsigned short* ap = &XT[(wave * 16 + l15) * SGT];
    af.h[0] = *(const us8*)(ap + 16 * half);
    af.h[1] = *(const us8*)(ap + 16 * half + 8);
    Frag16 bfr[2];
    load_bfrag(bfr[0], &YT[l15 * SGT + 16 * half]);
#pragma unroll
    for (int t = 0; t < 8; ++t) {
      if (t < 7) load_bfrag(bfr[(t + 1) & 1], &YT[((t + 1) * 16 + l15) * SGT + 16 * half]);
      acc[t] = wmma_bf16(af, bfr[t & 1], acc[t]);
    }
  }
  float* g = Gpart + ((size_t)b * gridDim.y + chunk) * 16384;
#pragma unroll
  for (int t = 0; t < 8; ++t) {
#pragma unroll
    for (int v = 0; v < 8; ++v) {
      const int m = wave * 16 + half * 8 + v;
      g[(size_t)m * 128 + (t * 16 + l15)] = acc[t][v];
    }
  }
}

// GT[b][n][k] = bf16( sum_c Gpart[b,c][k][n] ), k-pairs packed as u32.
__global__ __launch_bounds__(256)
void k_gram_reduce(const float* __restrict__ Gpart, unsigned* __restrict__ GT, int nc) {
  const int i = blockIdx.x * 256 + threadIdx.x;
  const int b = i >> 13, r = i & 8191;
  const int mp = r >> 7, n = r & 127, m = mp << 1;
  float s0 = 0.f, s1 = 0.f;
  for (int c = 0; c < nc; ++c) {
    const float* g = Gpart + ((size_t)b * nc + c) * 16384;
    s0 += g[(size_t)m * 128 + n];
    s1 += g[(size_t)(m + 1) * 128 + n];
  }
  GT[(size_t)b * 8192 + n * 64 + mp] = pk2bf(s0, s1);
}

// ---------------------------------------------------------------------------
// Per-batch context: out[b] = diag(rmask) * (A_bf[b][L,128] @ G[b]).
// ---------------------------------------------------------------------------
__global__ __launch_bounds__(128)
void k_gemm_ctx(const unsigned short* __restrict__ A, const unsigned short* __restrict__ GT,
                const int* __restrict__ rmask, float* __restrict__ out, int L) {
  __shared__ unsigned short BT[128 * SBT];
  const int b = blockIdx.y;
  const int tid = threadIdx.x, lane = tid & 31, wave = tid >> 5;
  const int half = lane >> 4, l15 = lane & 15;

  stage_bt(BT, GT + (size_t)b * 16384, tid);

  const int row0 = blockIdx.x * 128 + wave * 16;
  const unsigned short* arow0 = A + ((size_t)b * L + row0 + l15) * 128;
  const unsigned short* arow1 = arow0 + 64 * 128;

  v8f acc0[8] = {}, acc1[8] = {};
#pragma unroll
  for (int kk = 0; kk < 128; kk += 32) {
    Frag16 af0, af1;
    load_afrag_bf(af0, arow0 + kk, half);
    load_afrag_bf(af1, arow1 + kk, half);
    Frag16 bfr[2];
    load_bfrag(bfr[0], &BT[l15 * SBT + kk + 16 * half]);
#pragma unroll
    for (int t = 0; t < 8; ++t) {
      if (t < 7) load_bfrag(bfr[(t + 1) & 1], &BT[((t + 1) * 16 + l15) * SBT + kk + 16 * half]);
      acc0[t] = wmma_bf16(af0, bfr[t & 1], acc0[t]);
      acc1[t] = wmma_bf16(af1, bfr[t & 1], acc1[t]);
    }
  }
#pragma unroll
  for (int t = 0; t < 8; ++t) {
    const int col = t * 16 + l15;
#pragma unroll
    for (int v = 0; v < 8; ++v) {
      const int m = row0 + half * 8 + v;
      const float rm0 = (float)rmask[b * L + m];
      const float rm1 = (float)rmask[b * L + m + 64];
      out[((size_t)b * L + m) * 128 + col] = acc0[t][v] * rm0;
      out[((size_t)b * L + m + 64) * 128 + col] = acc1[t][v] * rm1;
    }
  }
}

// ---------------------------------------------------------------------------
__global__ __launch_bounds__(256)
void k_attn_pool(const float* __restrict__ ctx, const float* __restrict__ proj,
                 const int* __restrict__ mask, const float* __restrict__ Wa,
                 const float* __restrict__ ba, float* __restrict__ pooled, int L) {
  __shared__ float sbuf[4096];
  __shared__ float part[128];
  __shared__ float red[16];
  const int b = blockIdx.x, tid = threadIdx.x;
  const int warp = tid >> 5, lane = tid & 31;
  const size_t base = (size_t)b * L;

  for (int q = warp; q < L; q += 8) {
    const float* c = ctx + (base + q) * 128;
    float d = c[lane] * Wa[lane] + c[lane + 32] * Wa[lane + 32] +
              c[lane + 64] * Wa[lane + 64] + c[lane + 96] * Wa[lane + 96];
    for (int off = 16; off > 0; off >>= 1) d += __shfl_down(d, off, 32);
    if (lane == 0) sbuf[q] = d + ba[0];
  }
  __syncthreads();

  float mx = -3.4e38f;
  for (int q = tid; q < L; q += 256) mx = fmaxf(mx, sbuf[q]);
  for (int off = 16; off > 0; off >>= 1) mx = fmaxf(mx, __shfl_down(mx, off, 32));
  if (lane == 0) red[warp] = mx;
  __syncthreads();
  if (tid == 0) {
    float m = red[0];
    for (int i = 1; i < 8; ++i) m = fmaxf(m, red[i]);
    red[0] = m;
  }
  __syncthreads();
  mx = red[0];
  __syncthreads();

  float sum = 0.f;
  for (int q = tid; q < L; q += 256) {
    const float e = __expf(sbuf[q] - mx) * (float)mask[base + q];
    sbuf[q] = e;
    sum += e;
  }
  for (int off = 16; off > 0; off >>= 1) sum += __shfl_down(sum, off, 32);
  __syncthreads();
  if (lane == 0) red[warp] = sum;
  __syncthreads();
  if (tid == 0) {
    float s2 = 0.f;
    for (int i = 0; i < 8; ++i) s2 += red[i];
    red[8] = s2;
  }
  __syncthreads();
  const float inv = 1.f / (red[8] + 1e-6f);

  const int g = tid >> 7, h = tid & 127;
  const int qs = g * (L >> 1), qe = qs + (L >> 1);
  float accp = 0.f;
  for (int q = qs; q < qe; ++q) accp += sbuf[q] * proj[(base + q) * 128 + h];
  if (g) part[h] = accp;
  __syncthreads();
  if (!g) pooled[b * 128 + h] = (accp + part[h]) * inv;
}

// ---------------------------------------------------------------------------
__global__ __launch_bounds__(128)
void k_combine(const float* __restrict__ pooled, const float* __restrict__ W,
               const float* __restrict__ bias, float* __restrict__ out) {
  const int b = blockIdx.x, n = threadIdx.x;
  float a = bias[n];
  for (int j = 0; j < 384; ++j) {
    const int layer = j >> 7, jj = j & 127;
    a += pooled[(layer * 16 + b) * 128 + jj] * W[j * 128 + n];
  }
  out[b * 128 + n] = a;
}

// ---------------------------------------------------------------------------
extern "C" void kernel_launch(void* const* d_in, const int* in_sizes, int n_in,
                              void* d_out, int out_size, void* d_ws, size_t ws_size,
                              hipStream_t stream) {
  (void)in_sizes; (void)n_in; (void)out_size; (void)ws_size;
  const int B = 16, Lc = 512, Lp = 4096, H = 128;
  const int NCc = 2, NCp = 8;

  const float* comp_feat = (const float*)d_in[0];
  const int*   comp_mask = (const int*)d_in[1];
  const float* prot_feat = (const float*)d_in[2];
  const int*   prot_mask = (const int*)d_in[3];
  const float* Wc   = (const float*)d_in[4];
  const float* bc   = (const float*)d_in[5];
  const float* Wp   = (const float*)d_in[6];
  const float* bp   = (const float*)d_in[7];
  const float* Wbl  = (const float*)d_in[8];
  const float* Wac  = (const float*)d_in[9];
  const float* bac  = (const float*)d_in[10];
  const float* Wap  = (const float*)d_in[11];
  const float* bap  = (const float*)d_in[12];
  const float* Wcc  = (const float*)d_in[13];
  const float* bcc  = (const float*)d_in[14];
  const float* Wcp  = (const float*)d_in[15];
  const float* bcp  = (const float*)d_in[16];

  float* out = (float*)d_out;
  float* comp_final = out;
  float* prot_final = out + B * H;
  float* ctx_c_out  = out + 2 * B * H;
  float* ctx_p_out  = ctx_c_out + (size_t)B * Lc * H;

  char* w = (char*)d_ws;
  unsigned short* comp_feat_bf = (unsigned short*)w; w += (size_t)B * Lc * H * 2;
  unsigned short* prot_feat_bf = (unsigned short*)w; w += (size_t)B * Lp * H * 2;
  unsigned short* WcT = (unsigned short*)w; w += (size_t)3 * H * H * 2;
  unsigned short* WpT = (unsigned short*)w; w += (size_t)3 * H * H * 2;
  unsigned short* WblT = (unsigned short*)w; w += (size_t)3 * H * H * 2;
  float*          comp_proj    = (float*)w; w += (size_t)B * Lc * H * 4;
  unsigned short* comp_proj_bf = (unsigned short*)w; w += (size_t)B * Lc * H * 2;
  float*          prot_proj    = (float*)w; w += (size_t)B * Lp * H * 4;
  unsigned short* prot_proj_bf = (unsigned short*)w; w += (size_t)B * Lp * H * 2;
  unsigned short* comp_bil_bf  = (unsigned short*)w; w += (size_t)B * Lc * H * 2;
  unsigned short* GcT = (unsigned short*)w; w += (size_t)B * H * H * 2;
  unsigned short* GpT = (unsigned short*)w; w += (size_t)B * H * H * 2;
  float* Gpart    = (float*)w; w += (size_t)B * NCp * H * H * 4;
  float* ctx_c_ws = (float*)w; w += (size_t)B * Lc * H * 4;
  float* ctx_p_ws = (float*)w; w += (size_t)B * Lp * H * 4;
  float* pooled_c = (float*)w; w += (size_t)3 * B * H * 4;
  float* pooled_p = (float*)w; w += (size_t)3 * B * H * 4;

  k_cvt_bf16<<<(B * Lc * H / 2) / 256, 256, 0, stream>>>(comp_feat, (unsigned*)comp_feat_bf);
  k_cvt_bf16<<<(B * Lp * H / 2) / 256, 256, 0, stream>>>(prot_feat, (unsigned*)prot_feat_bf);
  k_wt_bf16<<<3, 256, 0, stream>>>(Wc, (unsigned*)WcT);
  k_wt_bf16<<<3, 256, 0, stream>>>(Wp, (unsigned*)WpT);
  k_wt_bf16<<<3, 256, 0, stream>>>(Wbl, (unsigned*)WblT);

  for (int i = 0; i < 3; ++i) {
    float* cc = (i == 2) ? ctx_c_out : ctx_c_ws;
    float* pc = (i == 2) ? ctx_p_out : ctx_p_ws;

    k_gemm_proj<<<(B * Lc) / 128, 128, 0, stream>>>(comp_feat_bf, WcT + i * H * H, bc + i * H,
                                                    comp_proj, comp_proj_bf, 1);
    k_gemm_proj<<<(B * Lp) / 128, 128, 0, stream>>>(prot_feat_bf, WpT + i * H * H, bp + i * H,
                                                    prot_proj, prot_proj_bf, 1);
    k_gemm_proj<<<(B * Lc) / 128, 128, 0, stream>>>(comp_proj_bf, WblT + i * H * H, nullptr,
                                                    nullptr, comp_bil_bf, 0);

    k_gemm_gram<<<dim3(B, NCc), 256, 0, stream>>>(comp_bil_bf, comp_proj_bf, comp_mask,
                                                  Gpart, Lc, Lc / NCc);
    k_gram_reduce<<<(B * H * H / 2) / 256, 256, 0, stream>>>(Gpart, (unsigned*)GcT, NCc);
    k_gemm_gram<<<dim3(B, NCp), 256, 0, stream>>>(prot_proj_bf, prot_proj_bf, prot_mask,
                                                  Gpart, Lp, Lp / NCp);
    k_gram_reduce<<<(B * H * H / 2) / 256, 256, 0, stream>>>(Gpart, (unsigned*)GpT, NCp);

    k_gemm_ctx<<<dim3(Lc / 128, B), 128, 0, stream>>>(comp_bil_bf, GpT, comp_mask, cc, Lc);
    k_gemm_ctx<<<dim3(Lp / 128, B), 128, 0, stream>>>(prot_proj_bf, GcT, prot_mask, pc, Lp);

    k_attn_pool<<<B, 256, 0, stream>>>(cc, comp_proj, comp_mask, Wac + i * H, bac + i,
                                       pooled_c + (size_t)i * B * H, Lc);
    k_attn_pool<<<B, 256, 0, stream>>>(pc, prot_proj, prot_mask, Wap + i * H, bap + i,
                                       pooled_p + (size_t)i * B * H, Lp);
  }

  k_combine<<<B, 128, 0, stream>>>(pooled_c, Wcc, bcc, comp_final);
  k_combine<<<B, 128, 0, stream>>>(pooled_p, Wcp, bcp, prot_final);
}
